// CAML_39977555591512
// MI455X (gfx1250) — matrix operand order repaired and checked by reading, not compiled, not run on hardware
//
#include <hip/hip_runtime.h>
#include <hip/hip_bf16.h>

typedef __attribute__((ext_vector_type(16))) _Float16 v16h;
typedef __attribute__((ext_vector_type(8)))  _Float16 v8h;
typedef __attribute__((ext_vector_type(8)))  float    v8f;

#define B_    8
#define L_    2500
#define E_    100
#define F_    50
#define C_    8921
#define K_    10
#define PAD_  5
#define LP_   2501          // conv output length
#define LPAD_ 2528          // 79 * 32
#define FPAD_ 64
#define CPAD_ 8928          // 279 * 32
#define CT2_  279           // c tile-pairs of 32 rows
#define NCH_  79            // l chunks of 32

#define ZH_BYTES ((size_t)B_ * LPAD_ * FPAD_ * 2)   // Z, (l,f)-major, f16, padded
#define ZT_BYTES ((size_t)B_ * FPAD_ * LPAD_ * 2)   // Z, (f,l)-major, f16, padded
#define QH_BYTES ((size_t)CPAD_ * FPAD_ * 2)        // Q_w f16 padded

// ---------------------------------------------------------------- helpers ---

__device__ __forceinline__ v8f wmma16(v16h a, v16h b, v8f c) {
  return __builtin_amdgcn_wmma_f32_16x16x32_f16(false, a, false, b,
                                                (short)0, c, false, false);
}

// ------------------------------------------------------------ prep kernels --

__global__ void zero_ws_kernel(unsigned int* p, int n) {
  int i = blockIdx.x * blockDim.x + threadIdx.x;
  if (i < n) p[i] = 0u;
}

__global__ void fill_qh_kernel(const float* __restrict__ Qw,
                               _Float16* __restrict__ Qh) {
  int i = blockIdx.x * blockDim.x + threadIdx.x;
  if (i < C_ * F_) {
    int c = i / F_, f = i % F_;
    Qh[(size_t)c * FPAD_ + f] = (_Float16)Qw[i];
  }
}

__global__ void fill_wt_kernel(const float* __restrict__ W,
                               float* __restrict__ WT) {
  int i = blockIdx.x * blockDim.x + threadIdx.x;
  if (i < F_ * E_ * K_) {
    int f = i / (E_ * K_);
    int r = i % (E_ * K_);
    int e = r / K_, k = r % K_;
    WT[(k * E_ + e) * F_ + f] = W[i];   // (k,e,f) layout, coalesced in f
  }
}

// embed -> conv1d -> tanh -> Z (f16, two layouts). 4 output positions/block.
__global__ __launch_bounds__(256) void conv_embed_kernel(
    const int* __restrict__ text, const float* __restrict__ embW,
    const float* __restrict__ convB, const float* __restrict__ convWT,
    _Float16* __restrict__ Zh, _Float16* __restrict__ Zt) {
  const int TL = 4;
  const int BPB = (LP_ + TL - 1) / TL;             // 626 blocks per batch
  int b  = blockIdx.x / BPB;
  int l0 = (blockIdx.x % BPB) * TL;

  __shared__ float xs[TL - 1 + K_][E_];            // 13 x 100 floats
  int tid = threadIdx.x;
  for (int idx = tid; idx < (TL - 1 + K_) * E_; idx += 256) {
    int r = idx / E_, e = idx % E_;
    int t = l0 + r - PAD_;
    float v = 0.f;
    if (t >= 0 && t < L_) {
      int tok = text[b * L_ + t];
      v = embW[(size_t)tok * E_ + e];               // padding_idx row is zero
    }
    xs[r][e] = v;
  }
  __syncthreads();

  int ll = tid >> 6;                                // 0..3
  int f  = tid & 63;
  int l  = l0 + ll;
  if (f < F_ && l < LP_) {
    float acc = convB[f];
    #pragma unroll
    for (int k = 0; k < K_; ++k) {
      const float* xr = xs[ll + k];
      const float* wr = convWT + k * E_ * F_ + f;
      for (int e = 0; e < E_; ++e)
        acc += xr[e] * wr[e * F_];
    }
    float z = tanhf(acc);
    Zh[((size_t)b * LPAD_ + l) * FPAD_ + f] = (_Float16)z;
    Zt[((size_t)b * FPAD_ + f) * LPAD_ + l] = (_Float16)z;
  }
}

// ------------------------------------------------------------- main kernel --
// One wave per (batch, 32-row c tile-pair): every Z B-operand load feeds two
// WMMA A-tiles, halving L2 read traffic. Pass 1: scores + online softmax
// stats. Pass 2: recompute scores, stream normalized alpha out with
// non-temporal stores, transpose through LDS, accumulate Ecf with WMMA.
__global__ __launch_bounds__(32) void caml_attn_kernel(
    const _Float16* __restrict__ Zh, const _Float16* __restrict__ Zt,
    const _Float16* __restrict__ Qh, const float* __restrict__ outW,
    const float* __restrict__ outB, float* __restrict__ logits,
    float* __restrict__ alpha) {
  const int lane = threadIdx.x;
  const int b  = blockIdx.x / CT2_;
  const int c0 = (blockIdx.x % CT2_) * 32;         // rows c0..c0+31 (2 tiles)
  const int m  = lane & 15;     // A-layout row / B,D-layout column
  const int kh = lane >> 4;     // K-half select

  // ---- A operands: two Q tiles (16 x 64), two K=32 slices each ----
  v16h aq[2][2];
  #pragma unroll
  for (int nt = 0; nt < 2; ++nt) {
    const _Float16* qrow = Qh + (size_t)(c0 + 16 * nt + m) * FPAD_;
    union { v16h v; _Float16 h[16]; } a0, a1;
    v8h lo0 = *(const v8h*)(qrow +      16 * kh);
    v8h hi0 = *(const v8h*)(qrow +  8 + 16 * kh);
    v8h lo1 = *(const v8h*)(qrow + 32 + 16 * kh);
    v8h hi1 = *(const v8h*)(qrow + 40 + 16 * kh);
    #pragma unroll
    for (int j = 0; j < 8; ++j) {
      a0.h[j] = lo0[j]; a0.h[j + 8] = hi0[j];
      a1.h[j] = lo1[j]; a1.h[j + 8] = hi1[j];
    }
    aq[nt][0] = a0.v; aq[nt][1] = a1.v;
  }

  const _Float16* ZhB = Zh + (size_t)b * LPAD_ * FPAD_;
  const v8f z8 = {0.f, 0.f, 0.f, 0.f, 0.f, 0.f, 0.f, 0.f};

  // ---- Pass 1: scores + per-lane online (max, sum) ----
  float rmax[2][8], rsum[2][8];
  #pragma unroll
  for (int nt = 0; nt < 2; ++nt)
    #pragma unroll
    for (int j = 0; j < 8; ++j) { rmax[nt][j] = -3.4e38f; rsum[nt][j] = 0.f; }

  for (int ch = 0; ch < NCH_; ++ch) {
    int l0 = ch * 32;
    #pragma unroll
    for (int t2 = 0; t2 < 2; ++t2) {
      int l0t = l0 + 16 * t2;
      const _Float16* zr = ZhB + (size_t)(l0t + m) * FPAD_ + 16 * kh;
      v16h b0 = *(const v16h*)(zr);
      v16h b1 = *(const v16h*)(zr + 32);
      union { v8f v; float f[8]; } d[2];
      #pragma unroll
      for (int nt = 0; nt < 2; ++nt) {
        d[nt].v = wmma16(aq[nt][0], b0, z8);
        d[nt].v = wmma16(aq[nt][1], b1, d[nt].v);
      }
      if (l0t + m < LP_) {
        #pragma unroll
        for (int nt = 0; nt < 2; ++nt)
          #pragma unroll
          for (int j = 0; j < 8; ++j) {
            float s  = d[nt].f[j];
            float mo = rmax[nt][j];
            float mn = fmaxf(mo, s);
            rsum[nt][j] = rsum[nt][j] * __expf(mo - mn) + __expf(s - mn);
            rmax[nt][j] = mn;
          }
      }
    }
  }
  // combine across the 16 lanes that share each row (xor tree within halves)
  #pragma unroll
  for (int off = 1; off < 16; off <<= 1)
    #pragma unroll
    for (int nt = 0; nt < 2; ++nt)
      #pragma unroll
      for (int j = 0; j < 8; ++j) {
        float om = __shfl_xor(rmax[nt][j], off, 32);
        float os = __shfl_xor(rsum[nt][j], off, 32);
        float mn = fmaxf(rmax[nt][j], om);
        rsum[nt][j] = rsum[nt][j] * __expf(rmax[nt][j] - mn)
                    + os * __expf(om - mn);
        rmax[nt][j] = mn;
      }
  float rinv[2][8];
  #pragma unroll
  for (int nt = 0; nt < 2; ++nt)
    #pragma unroll
    for (int j = 0; j < 8; ++j) rinv[nt][j] = 1.0f / rsum[nt][j];

  // ---- Pass 2: recompute scores -> alpha out + Ecf WMMA accumulation ----
  __shared__ _Float16 ls[2][16][32];               // alpha staging (D -> A)
  v8f cE[2][4];
  #pragma unroll
  for (int nt = 0; nt < 2; ++nt)
    #pragma unroll
    for (int t = 0; t < 4; ++t) cE[nt][t] = z8;

  for (int ch = 0; ch < NCH_; ++ch) {
    int l0 = ch * 32;
    #pragma unroll
    for (int t2 = 0; t2 < 2; ++t2) {
      int l0t = l0 + 16 * t2;
      const _Float16* zr = ZhB + (size_t)(l0t + m) * FPAD_ + 16 * kh;
      v16h b0 = *(const v16h*)(zr);
      v16h b1 = *(const v16h*)(zr + 32);
      union { v8f v; float f[8]; } d[2];
      #pragma unroll
      for (int nt = 0; nt < 2; ++nt) {
        d[nt].v = wmma16(aq[nt][0], b0, z8);
        d[nt].v = wmma16(aq[nt][1], b1, d[nt].v);
      }
      int  l    = l0t + m;
      int  lloc = 16 * t2 + m;
      bool lok  = (l < LP_);
      #pragma unroll
      for (int nt = 0; nt < 2; ++nt)
        #pragma unroll
        for (int j = 0; j < 8; ++j) {
          float av = lok ? __expf(d[nt].f[j] - rmax[nt][j]) * rinv[nt][j] : 0.f;
          ls[nt][j + 8 * kh][lloc] = (_Float16)av;
          int c = c0 + 16 * nt + j + 8 * kh;
          if (lok && c < C_)   // write-once stream: keep it out of L2 (NT hint)
            __builtin_nontemporal_store(av,
                &alpha[((size_t)b * C_ + c) * LP_ + l]);
        }
    }
    __syncthreads();
    // read alpha back in A layout: elem j -> K(col) = j+16*kh
    union { v16h v; _Float16 h[16]; } aa[2];
    #pragma unroll
    for (int nt = 0; nt < 2; ++nt)
      #pragma unroll
      for (int j = 0; j < 16; ++j)
        aa[nt].h[j] = ls[nt][m][j + 16 * kh];
    #pragma unroll
    for (int tf = 0; tf < 4; ++tf) {
      const _Float16* ztr =
          Zt + ((size_t)b * FPAD_ + tf * 16 + m) * LPAD_ + l0 + 16 * kh;
      v16h bz = *(const v16h*)ztr;                 // contiguous 32B, B layout
      cE[0][tf] = wmma16(aa[0].v, bz, cE[0][tf]);
      cE[1][tf] = wmma16(aa[1].v, bz, cE[1][tf]);
    }
    __syncthreads();
  }

  // ---- logits: per-class dot of Ecf rows with out_w ----
  #pragma unroll
  for (int nt = 0; nt < 2; ++nt) {
    union { v8f v; float f[8]; } eU[4];
    #pragma unroll
    for (int t = 0; t < 4; ++t) eU[t].v = cE[nt][t];
    #pragma unroll
    for (int j = 0; j < 8; ++j) {
      int c = c0 + 16 * nt + j + 8 * kh;
      float p = 0.f;
      if (c < C_) {
        #pragma unroll
        for (int tf = 0; tf < 4; ++tf) {
          int f = tf * 16 + m;
          if (f < F_) p += outW[(size_t)c * F_ + f] * eU[tf].f[j];
        }
      }
      #pragma unroll
      for (int off = 1; off < 16; off <<= 1) p += __shfl_xor(p, off, 32);
      if (m == 0 && c < C_) logits[(size_t)b * C_ + c] = p + outB[c];
    }
  }
}

// ---------------------------------------------------------------- launcher --

extern "C" void kernel_launch(void* const* d_in, const int* in_sizes, int n_in,
                              void* d_out, int out_size, void* d_ws,
                              size_t ws_size, hipStream_t stream) {
  const int*   text  = (const int*)d_in[0];
  const float* embW  = (const float*)d_in[1];
  const float* convW = (const float*)d_in[2];
  const float* convB = (const float*)d_in[3];
  const float* Qw    = (const float*)d_in[4];
  const float* outW  = (const float*)d_in[5];
  const float* outB  = (const float*)d_in[6];

  float* logits = (float*)d_out;
  float* alpha  = logits + (size_t)B_ * C_;

  char* ws = (char*)d_ws;
  _Float16* Zh     = (_Float16*)(ws);
  _Float16* Zt     = (_Float16*)(ws + ZH_BYTES);
  _Float16* Qh     = (_Float16*)(ws + ZH_BYTES + ZT_BYTES);
  float*    convWT = (float*)(ws + ZH_BYTES + ZT_BYTES + QH_BYTES);

  // zero the f16 padded regions (padding lanes must be exactly 0)
  int nz = (int)((ZH_BYTES + ZT_BYTES + QH_BYTES) / 4);
  zero_ws_kernel<<<(nz + 255) / 256, 256, 0, stream>>>((unsigned int*)ws, nz);

  fill_qh_kernel<<<(C_ * F_ + 255) / 256, 256, 0, stream>>>(Qw, Qh);
  fill_wt_kernel<<<(F_ * E_ * K_ + 255) / 256, 256, 0, stream>>>(convW, convWT);

  conv_embed_kernel<<<B_ * ((LP_ + 3) / 4), 256, 0, stream>>>(
      text, embW, convB, convWT, Zh, Zt);

  caml_attn_kernel<<<B_ * CT2_, 32, 0, stream>>>(
      Zh, Zt, Qh, outW, outB, logits, alpha);
}
